// GRUCell_27719718928937
// MI455X (gfx1250) — matrix operand order repaired and checked by reading, not compiled, not run on hardware
//
#include <hip/hip_runtime.h>
#include <hip/hip_bf16.h>
#include <math.h>

#define BATCH 8192
#define HID   2048
#define KTOT  4096

#define BM  128
#define BN  64
#define KB  64
#define LDT 64    // LDS row stride in bf16 elements (128 B rows, unpadded)

typedef __bf16 bf16;
typedef __attribute__((ext_vector_type(16))) __bf16 v16bf;
typedef __attribute__((ext_vector_type(8)))  __bf16 v8bf;
typedef __attribute__((ext_vector_type(4)))  __bf16 v4bf;
typedef __attribute__((ext_vector_type(8)))  float  v8f;
typedef __attribute__((ext_vector_type(4)))  int    v4i;

typedef __attribute__((address_space(1))) v4i gv4i;   // global int4
typedef __attribute__((address_space(3))) v4i lv4i;   // LDS int4

union V16U { v16bf v; v8bf h[2]; };

// ---------------- async global->LDS (ASYNCcnt path, gfx1250) ----------------
__device__ __forceinline__ void async_b128(bf16* lds_ptr, const bf16* g_ptr) {
#if __has_builtin(__builtin_amdgcn_global_load_async_to_lds_b128)
  __builtin_amdgcn_global_load_async_to_lds_b128(
      (gv4i*)g_ptr, (lv4i*)lds_ptr, 0, 0);
#else
  const unsigned l = (unsigned)(unsigned long long)lds_ptr;
  asm volatile("global_load_async_to_lds_b128 %0, %1, off"
               :: "v"(l), "v"(g_ptr) : "memory");
#endif
}

__device__ __forceinline__ void wait_async0() {
#if __has_builtin(__builtin_amdgcn_s_wait_asynccnt)
  __builtin_amdgcn_s_wait_asynccnt(0);
#else
  asm volatile("s_wait_asynccnt 0x0" ::: "memory");
#endif
}

// ---------------- WMMA fragment loads from LDS ----------------
// A fragment (16x32 bf16): lane l holds M=l&15, K in {kh..kh+7} U {kh+16..kh+23},
// kh = (l>=16)*8 (CDNA5 16-bit A layout).
__device__ __forceinline__ v16bf lds_a_frag(const bf16* As, int lane, int mbase, int koff) {
  const int m  = mbase + (lane & 15);
  const int kh = koff + (lane >> 4) * 8;
  V16U u;
  u.h[0] = *reinterpret_cast<const v8bf*>(As + m * LDT + kh);
  u.h[1] = *reinterpret_cast<const v8bf*>(As + m * LDT + kh + 16);
  return u.v;
}

// B fragment (32x16 bf16): lane l holds N=l&15, K in {kb..kb+7} U {kb+16..kb+23}, kb = koff+(l>=16)*8.
// Bs holds the tile transposed: Bs[n][k].
__device__ __forceinline__ v16bf lds_b_frag(const bf16* Bs, int lane, int nbase, int koff) {
  const int n  = nbase + (lane & 15);
  const int kb = koff + (lane >> 4) * 8;
  V16U u;
  u.h[0] = *reinterpret_cast<const v8bf*>(Bs + n * LDT + kb);
  u.h[1] = *reinterpret_cast<const v8bf*>(Bs + n * LDT + kb + 16);
  return u.v;
}

// ---------------- async tile staging (pure bf16 byte copies) ----------------
// A tile: BM x KB bf16 (src row stride HID). 128 rows * 8 x 16B chunks = 4/thread.
__device__ __forceinline__ void stage_a(bf16* dst, const bf16* __restrict__ src,
                                        int m0, int kofs, int tid) {
#pragma unroll
  for (int i = 0; i < 4; ++i) {
    const int idx = tid + i * 256;     // 0..1023
    const int row = idx >> 3;          // 0..127
    const int ch  = (idx & 7) * 8;     // bf16 offset, 16B chunks
    async_b128(dst + row * LDT + ch,
               src + (size_t)(m0 + row) * HID + kofs + ch);
  }
}

// B tile: BN x KB bf16 from N-major transposed weights (row stride KTOT). 2/thread.
__device__ __forceinline__ void stage_b(bf16* dst, const bf16* __restrict__ wt,
                                        int n0, int k0, int tid) {
#pragma unroll
  for (int i = 0; i < 2; ++i) {
    const int idx = tid + i * 256;     // 0..511
    const int row = idx >> 3;          // 0..63
    const int ch  = (idx & 7) * 8;
    async_b128(dst + row * LDT + ch,
               wt + (size_t)(n0 + row) * KTOT + k0 + ch);
  }
}

#define WMMA_BF16(A, B, C) \
  __builtin_amdgcn_wmma_f32_16x16x32_bf16(false, (A), false, (B), (short)0, (C), false, false)

// ---------------- precision-conversion prepass ----------------
__global__ __launch_bounds__(256) void f2bf_kernel(const float* __restrict__ in,
                                                   bf16* __restrict__ out) {
  const int i = blockIdx.x * 256 + threadIdx.x;         // one float4 each
  const float4 f = reinterpret_cast<const float4*>(in)[i];
  reinterpret_cast<v4bf*>(out)[i] = v4bf{(bf16)f.x, (bf16)f.y, (bf16)f.z, (bf16)f.w};
}

// Convert W (KTOT x HID f32, K-major) -> WT (HID x KTOT bf16, N-major) via LDS tiles.
#define TDIM 64
#define TLD  72
__global__ __launch_bounds__(256) void wconv_kernel(const float* __restrict__ W,
                                                    bf16* __restrict__ WT) {
  __shared__ bf16 T[TDIM * TLD];                        // T[k][n]
  const int tid = threadIdx.x;
  const int k0 = blockIdx.y * TDIM;
  const int n0 = blockIdx.x * TDIM;
  {
    const int kr  = tid >> 2;                           // 0..63
    const int c16 = (tid & 3) * 16;
#pragma unroll
    for (int q = 0; q < 4; ++q) {
      const float4 f = *reinterpret_cast<const float4*>(
          W + (size_t)(k0 + kr) * HID + n0 + c16 + q * 4);
      *reinterpret_cast<v4bf*>(&T[kr * TLD + c16 + q * 4]) =
          v4bf{(bf16)f.x, (bf16)f.y, (bf16)f.z, (bf16)f.w};
    }
  }
  __syncthreads();
  {
    const int nr = tid >> 2;                            // 0..63
    const int ks = (tid & 3) * 16;
    alignas(16) bf16 tmp[16];
#pragma unroll
    for (int q = 0; q < 16; ++q) tmp[q] = T[(ks + q) * TLD + nr];
    bf16* o = WT + (size_t)(n0 + nr) * KTOT + k0 + ks;
    *reinterpret_cast<v8bf*>(o)     = *reinterpret_cast<v8bf*>(&tmp[0]);
    *reinterpret_cast<v8bf*>(o + 8) = *reinterpret_cast<v8bf*>(&tmp[8]);
  }
}

// ---------------- kernel 1: z and r gates ----------------
__global__ __launch_bounds__(256) void gru_zr_kernel(
    const bf16* __restrict__ x16, const bf16* __restrict__ h16,
    const bf16* __restrict__ WzT, const bf16* __restrict__ WrT,
    const float* __restrict__ bz,  const float* __restrict__ br,
    const float* __restrict__ h_prev,
    bf16* __restrict__ z16, bf16* __restrict__ rh16) {
  __shared__ bf16 As [2][BM * LDT];
  __shared__ bf16 Bsz[2][BN * LDT];
  __shared__ bf16 Bsr[2][BN * LDT];

  const int tid   = threadIdx.x;
  const int lane  = tid & 31;
  const int wave  = tid >> 5;
  const int waveM = (wave >> 1) * 32;
  const int waveN = (wave & 1) * 32;
  const int m0 = blockIdx.y * BM;
  const int n0 = blockIdx.x * BN;

  v8f accz[2][2] = {};
  v8f accr[2][2] = {};

  // prologue: stage k-block 0
  stage_a(As[0], x16, m0, 0, tid);
  stage_b(Bsz[0], WzT, n0, 0, tid);
  stage_b(Bsr[0], WrT, n0, 0, tid);

  const int NK = KTOT / KB;                             // 64
  for (int kb = 0; kb < NK; ++kb) {
    const int cur = kb & 1;
    wait_async0();
    __syncthreads();                                    // buf[cur] ready; buf[cur^1] free

    const int knxt = (kb + 1) * KB;
    if (knxt < KTOT) {
      const int   nxt  = cur ^ 1;
      const bf16* Asrc = (knxt < HID) ? x16 : h16;
      const int   kofs = (knxt < HID) ? knxt : (knxt - HID);
      stage_a(As[nxt], Asrc, m0, kofs, tid);
      stage_b(Bsz[nxt], WzT, n0, knxt, tid);
      stage_b(Bsr[nxt], WrT, n0, knxt, tid);
    }

#pragma unroll
    for (int s = 0; s < 2; ++s) {
      const int ko = s * 32;
      const v16bf a0  = lds_a_frag(As[cur],  lane, waveM,      ko);
      const v16bf a1  = lds_a_frag(As[cur],  lane, waveM + 16, ko);
      const v16bf bz0 = lds_b_frag(Bsz[cur], lane, waveN,      ko);
      const v16bf bz1 = lds_b_frag(Bsz[cur], lane, waveN + 16, ko);
      const v16bf br0 = lds_b_frag(Bsr[cur], lane, waveN,      ko);
      const v16bf br1 = lds_b_frag(Bsr[cur], lane, waveN + 16, ko);
      accz[0][0] = WMMA_BF16(a0, bz0, accz[0][0]);
      accz[0][1] = WMMA_BF16(a0, bz1, accz[0][1]);
      accz[1][0] = WMMA_BF16(a1, bz0, accz[1][0]);
      accz[1][1] = WMMA_BF16(a1, bz1, accz[1][1]);
      accr[0][0] = WMMA_BF16(a0, br0, accr[0][0]);
      accr[0][1] = WMMA_BF16(a0, br1, accr[0][1]);
      accr[1][0] = WMMA_BF16(a1, br0, accr[1][0]);
      accr[1][1] = WMMA_BF16(a1, br1, accr[1][1]);
    }
  }

  // epilogue: C layout -> M = j + (lane>>4)*8, N = lane&15 per 16x16 block
#pragma unroll
  for (int mi = 0; mi < 2; ++mi) {
#pragma unroll
    for (int ni = 0; ni < 2; ++ni) {
      const int col  = n0 + waveN + ni * 16 + (lane & 15);
      const int rowb = m0 + waveM + mi * 16 + (lane >> 4) * 8;
      const float bzc = bz[col];
      const float brc = br[col];
#pragma unroll
      for (int j = 0; j < 8; ++j) {
        const size_t idx = (size_t)(rowb + j) * HID + col;
        const float zc = 1.0f / (1.0f + __expf(-(accz[mi][ni][j] + bzc)));
        const float rc = 1.0f / (1.0f + __expf(-(accr[mi][ni][j] + brc)));
        z16[idx]  = (bf16)zc;
        rh16[idx] = (bf16)(rc * h_prev[idx]);
      }
    }
  }
}

// ---------------- kernel 2: n gate + GRU output ----------------
__global__ __launch_bounds__(256) void gru_n_kernel(
    const bf16* __restrict__ x16, const bf16* __restrict__ rh16,
    const bf16* __restrict__ WnT, const float* __restrict__ bn,
    const bf16* __restrict__ z16, const float* __restrict__ h_prev,
    float* __restrict__ out) {
  __shared__ bf16 As [2][BM * LDT];
  __shared__ bf16 Bsn[2][BN * LDT];

  const int tid   = threadIdx.x;
  const int lane  = tid & 31;
  const int wave  = tid >> 5;
  const int waveM = (wave >> 1) * 32;
  const int waveN = (wave & 1) * 32;
  const int m0 = blockIdx.y * BM;
  const int n0 = blockIdx.x * BN;

  v8f accn[2][2] = {};

  stage_a(As[0], x16, m0, 0, tid);
  stage_b(Bsn[0], WnT, n0, 0, tid);

  const int NK = KTOT / KB;
  for (int kb = 0; kb < NK; ++kb) {
    const int cur = kb & 1;
    wait_async0();
    __syncthreads();

    const int knxt = (kb + 1) * KB;
    if (knxt < KTOT) {
      const int   nxt  = cur ^ 1;
      const bf16* Asrc = (knxt < HID) ? x16 : rh16;
      const int   kofs = (knxt < HID) ? knxt : (knxt - HID);
      stage_a(As[nxt], Asrc, m0, kofs, tid);
      stage_b(Bsn[nxt], WnT, n0, knxt, tid);
    }

#pragma unroll
    for (int s = 0; s < 2; ++s) {
      const int ko = s * 32;
      const v16bf a0  = lds_a_frag(As[cur],  lane, waveM,      ko);
      const v16bf a1  = lds_a_frag(As[cur],  lane, waveM + 16, ko);
      const v16bf bn0 = lds_b_frag(Bsn[cur], lane, waveN,      ko);
      const v16bf bn1 = lds_b_frag(Bsn[cur], lane, waveN + 16, ko);
      accn[0][0] = WMMA_BF16(a0, bn0, accn[0][0]);
      accn[0][1] = WMMA_BF16(a0, bn1, accn[0][1]);
      accn[1][0] = WMMA_BF16(a1, bn0, accn[1][0]);
      accn[1][1] = WMMA_BF16(a1, bn1, accn[1][1]);
    }
  }

#pragma unroll
  for (int mi = 0; mi < 2; ++mi) {
#pragma unroll
    for (int ni = 0; ni < 2; ++ni) {
      const int col  = n0 + waveN + ni * 16 + (lane & 15);
      const int rowb = m0 + waveM + mi * 16 + (lane >> 4) * 8;
      const float bnc = bn[col];
#pragma unroll
      for (int j = 0; j < 8; ++j) {
        const size_t idx = (size_t)(rowb + j) * HID + col;
        const float nv = tanhf(accn[mi][ni][j] + bnc);
        const float zv = (float)z16[idx];
        const float hp = h_prev[idx];
        out[idx] = (1.0f - zv) * hp + zv * nv;
      }
    }
  }
}

// ---------------- host launch ----------------
extern "C" void kernel_launch(void* const* d_in, const int* in_sizes, int n_in,
                              void* d_out, int out_size, void* d_ws, size_t ws_size,
                              hipStream_t stream) {
  const float* x  = (const float*)d_in[0];
  const float* h  = (const float*)d_in[1];
  const float* Wz = (const float*)d_in[2];
  const float* bz = (const float*)d_in[3];
  const float* Wr = (const float*)d_in[4];
  const float* br = (const float*)d_in[5];
  const float* Wn = (const float*)d_in[6];
  const float* bn = (const float*)d_in[7];
  float* out = (float*)d_out;

  const size_t NB = (size_t)BATCH * HID;     // 16.7M elements
  const size_t WN = (size_t)HID * KTOT;      //  8.4M elements
  bf16* x16  = (bf16*)d_ws;
  bf16* h16  = x16  + NB;
  bf16* rh16 = h16  + NB;
  bf16* z16  = rh16 + NB;
  bf16* Wz16 = z16  + NB;
  bf16* Wr16 = Wz16 + WN;
  bf16* Wn16 = Wr16 + WN;                    // total ~184 MB

  // precision prepass: bf16 operands + N-major weights
  f2bf_kernel<<<(int)(NB / 1024), 256, 0, stream>>>(x, x16);
  f2bf_kernel<<<(int)(NB / 1024), 256, 0, stream>>>(h, h16);
  dim3 wgrid(HID / TDIM, KTOT / TDIM);       // 32 x 64
  wconv_kernel<<<wgrid, 256, 0, stream>>>(Wz, Wz16);
  wconv_kernel<<<wgrid, 256, 0, stream>>>(Wr, Wr16);
  wconv_kernel<<<wgrid, 256, 0, stream>>>(Wn, Wn16);

  dim3 grid(HID / BN, BATCH / BM);           // 32 x 64
  gru_zr_kernel<<<grid, 256, 0, stream>>>(x16, h16, Wz16, Wr16, bz, br, h, z16, rh16);
  gru_n_kernel <<<grid, 256, 0, stream>>>(x16, rh16, Wn16, bn, z16, h, out);
}
